// DecoderModel_85581518340683
// MI455X (gfx1250) — compile-verified, matrix operand order checked by reading, not run on hardware
//
#include <hip/hip_runtime.h>
#include <hip/hip_bf16.h>

// ---------------------------------------------------------------------------
// GPT decoder forward for gfx1250 (MI455X): bf16 WMMA GEMMs + flash attention.
// Dims fixed by the reference: B=4,T=1024,E=1024,H=16,hs=64,L=4,F=4096,V=32000
// ---------------------------------------------------------------------------

typedef __attribute__((ext_vector_type(16))) __bf16 v16bf;
typedef __attribute__((ext_vector_type(8)))  float  v8f;
typedef __attribute__((ext_vector_type(4)))  float  f32x4;

#define WMMA_BF16(a, b, c) \
    __builtin_amdgcn_wmma_f32_16x16x32_bf16(false, (a), false, (b), (short)0, (c), false, false)

// Fragment loader: elements 0..7 = p[kb..kb+7], 8..15 = p[kb+16..kb+23]
// (16-bit A/B matrix layout per CDNA5 ISA 7.12.2; caller supplies
//  kb = k0 + ((lane&16)?8:0) and p = base + row_or_col * ld)
__device__ __forceinline__ v16bf frag_ld(const __bf16* p, int kb) {
    union U { f32x4 f[2]; v16bf v; } u;
    u.f[0] = *reinterpret_cast<const f32x4*>(p + kb);
    u.f[1] = *reinterpret_cast<const f32x4*>(p + kb + 16);
    return u.v;
}

// ---------------------------------------------------------------------------
// Elementwise / conversion kernels
// ---------------------------------------------------------------------------

// x[b,t,e] = tok_emb[idx[b,t],e] + pos_emb[t,e]   (i decodes b|t|e, 10 bits each)
__global__ void embed_kernel(const int* __restrict__ idx, const float* __restrict__ tok,
                             const float* __restrict__ pos, float* __restrict__ x, int total) {
    int i = blockIdx.x * blockDim.x + threadIdx.x;
    if (i >= total) return;
    int e = i & 1023, t = (i >> 10) & 1023, b = i >> 20;
    x[i] = tok[(size_t)idx[b * 1024 + t] * 1024 + e] + pos[t * 1024 + e];
}

// Wq/Wk/Wv [L,H,E,hs] fp32 -> [L, H*hs, E] bf16 (transposed [N,K] layout)
__global__ void conv_qkvT_kernel(const float* __restrict__ W, __bf16* __restrict__ out, int total) {
    int i = blockIdx.x * blockDim.x + threadIdx.x;
    if (i >= total) return;
    int e = i & 1023, n = (i >> 10) & 1023, l = i >> 20;
    int h = n >> 6, d = n & 63;
    out[i] = (__bf16)W[(((size_t)(l * 16 + h) * 1024) + e) * 64 + d];
}

// W [batch,K,N] fp32 -> out [batch,N,K] bf16
__global__ void transpose_to_bf16_kernel(const float* __restrict__ W, __bf16* __restrict__ out,
                                         int K, int N, int total) {
    int i = blockIdx.x * blockDim.x + threadIdx.x;
    if (i >= total) return;
    int k = i % K;
    int nl = i / K;
    int n = nl % N;
    int l = nl / N;
    out[i] = (__bf16)W[(size_t)l * K * N + (size_t)k * N + n];
}

// v [B,T,E] bf16 -> vT [B,E,T] bf16
__global__ void vtrans_kernel(const __bf16* __restrict__ v, __bf16* __restrict__ out, int total) {
    int i = blockIdx.x * blockDim.x + threadIdx.x;
    if (i >= total) return;
    int t = i & 1023, e = (i >> 10) & 1023, b = i >> 20;
    out[i] = v[((size_t)(b * 1024 + t)) * 1024 + e];
}

// Row layernorm over E=1024, fp32 in -> bf16 out
__global__ void layernorm_kernel(const float* __restrict__ x, const float* __restrict__ g,
                                 const float* __restrict__ bb, __bf16* __restrict__ out) {
    constexpr int E = 1024;
    int row = blockIdx.x;
    const float* xr = x + (size_t)row * E;
    float s = 0.f, ss = 0.f;
    for (int i = threadIdx.x; i < E; i += blockDim.x) { float v = xr[i]; s += v; ss += v * v; }
    for (int off = 16; off > 0; off >>= 1) { s += __shfl_down(s, off); ss += __shfl_down(ss, off); }
    __shared__ float rs[8], rss[8];
    int wave = threadIdx.x >> 5, lane = threadIdx.x & 31;
    if (lane == 0) { rs[wave] = s; rss[wave] = ss; }
    __syncthreads();
    if (threadIdx.x == 0) {
        float a = 0.f, c = 0.f;
        for (int w = 0; w < 8; w++) { a += rs[w]; c += rss[w]; }
        rs[0] = a; rss[0] = c;
    }
    __syncthreads();
    float mu = rs[0] * (1.f / E);
    float var = rss[0] * (1.f / E) - mu * mu;
    float rstd = rsqrtf(var + 1e-5f);
    for (int i = threadIdx.x; i < E; i += blockDim.x)
        out[(size_t)row * E + i] = (__bf16)((xr[i] - mu) * rstd * g[i] + bb[i]);
}

// ---------------------------------------------------------------------------
// WMMA GEMM: C[M,N] = A[M,K] * Bt^T (+bias)(+residual)(relu?)
//   A  : bf16 [M,K] row-major
//   Bt : bf16 [N,K] (transposed weights)
//   block = 128 threads (4 waves), wave tile = 16(M) x 64(N), K step 32
//   All of M, N divisible by 64; K divisible by 32.
// ---------------------------------------------------------------------------
__global__ void gemm_bf16_kernel(const __bf16* __restrict__ A, const __bf16* __restrict__ Bt,
                                 const float* __restrict__ bias, const float* __restrict__ res,
                                 float* __restrict__ Cf, __bf16* __restrict__ Cb,
                                 int N, int K, int relu) {
    int lane = threadIdx.x & 31, wave = threadIdx.x >> 5;
    int l15 = lane & 15, hi8 = (lane & 16) ? 8 : 0;
    int tileM = (blockIdx.y * 4 + wave) * 16;
    int tileN = blockIdx.x * 64;
    const __bf16* arow = A + (size_t)(tileM + l15) * K;
    const __bf16* brow = Bt + (size_t)(tileN + l15) * K;
    v8f acc[4] = {};
    for (int k0 = 0; k0 < K; k0 += 32) {
        int kb = k0 + hi8;
        v16bf a  = frag_ld(arow, kb);
        v16bf w0 = frag_ld(brow, kb);
        v16bf w1 = frag_ld(brow + (size_t)16 * K, kb);
        v16bf w2 = frag_ld(brow + (size_t)32 * K, kb);
        v16bf w3 = frag_ld(brow + (size_t)48 * K, kb);
        acc[0] = WMMA_BF16(a, w0, acc[0]);
        acc[1] = WMMA_BF16(a, w1, acc[1]);
        acc[2] = WMMA_BF16(a, w2, acc[2]);
        acc[3] = WMMA_BF16(a, w3, acc[3]);
    }
#pragma unroll
    for (int s = 0; s < 4; s++) {
        int col = tileN + s * 16 + l15;
        float bv = bias ? bias[col] : 0.f;
#pragma unroll
        for (int r = 0; r < 8; r++) {
            int row = tileM + r + hi8;
            size_t o = (size_t)row * N + col;
            float v = acc[s][r] + bv;
            if (res) v += res[o];
            if (relu) v = fmaxf(v, 0.f);
            if (Cf) Cf[o] = v;
            if (Cb) Cb[o] = (__bf16)v;
        }
    }
}

// ---------------------------------------------------------------------------
// Flash attention (causal, online softmax). One wave per 16 query rows.
//   q,k : bf16 [B,T,E]   vT : bf16 [B,E,T]   att out : bf16 [B,T,E]
//   KV chunk = 32 tokens. Scores never touch HBM.
// ---------------------------------------------------------------------------
__global__ void attn_kernel(const __bf16* __restrict__ q, const __bf16* __restrict__ k,
                            const __bf16* __restrict__ vT, __bf16* __restrict__ att,
                            float scale) {
    constexpr int T = 1024, E = 1024, HS = 64;
    int lane = threadIdx.x & 31, wave = threadIdx.x >> 5;
    int l15 = lane & 15, hi8 = (lane & 16) ? 8 : 0;
    int h = blockIdx.y, b = blockIdx.z;
    int qbase = (blockIdx.x * 4 + wave) * 16;

    __shared__ __bf16 plds[4][16 * 40];   // per-wave 16x32 P tile, ld=40 (16B-aligned rows)
    __bf16* pl = &plds[wave][0];

    const __bf16* qrow = q + ((size_t)(b * T + qbase + l15)) * E + h * HS;
    v16bf aq0 = frag_ld(qrow, hi8);        // K = 0..31 of head dim
    v16bf aq1 = frag_ld(qrow, 32 + hi8);   // K = 32..63

    float m[8], l[8];
#pragma unroll
    for (int r = 0; r < 8; r++) { m[r] = -1e30f; l[r] = 0.f; }
    v8f o[4] = {};

    int nch = (qbase + 47) / 32;           // chunks of 32 keys covering rows <= qbase+15
    for (int j = 0; j < nch; j++) {
        int jj = j * 32;
        // S = Q * K^T for 32 keys -> two 16x16 tiles
        const __bf16* kr0 = k + ((size_t)(b * T + jj + l15)) * E + h * HS;
        const __bf16* kr1 = kr0 + (size_t)16 * E;
        v8f s0 = {}, s1 = {};
        s0 = WMMA_BF16(aq0, frag_ld(kr0, hi8), s0);
        s0 = WMMA_BF16(aq1, frag_ld(kr0, 32 + hi8), s0);
        s1 = WMMA_BF16(aq0, frag_ld(kr1, hi8), s1);
        s1 = WMMA_BF16(aq1, frag_ld(kr1, 32 + hi8), s1);

#pragma unroll
        for (int r = 0; r < 8; r++) {
            int row = qbase + r + hi8;
            float x0 = s0[r] * scale;
            float x1 = s1[r] * scale;
            if (jj + l15 > row)      x0 = -1e30f;   // causal mask
            if (jj + 16 + l15 > row) x1 = -1e30f;
            float mx = fmaxf(x0, x1);
            mx = fmaxf(mx, __shfl_xor(mx, 1));
            mx = fmaxf(mx, __shfl_xor(mx, 2));
            mx = fmaxf(mx, __shfl_xor(mx, 4));
            mx = fmaxf(mx, __shfl_xor(mx, 8));
            float mn = fmaxf(m[r], mx);
            float al = __expf(m[r] - mn);
            m[r] = mn;
            float p0 = __expf(x0 - mn);
            float p1 = __expf(x1 - mn);
            int prow = r + hi8;
            pl[prow * 40 + l15]      = (__bf16)p0;
            pl[prow * 40 + 16 + l15] = (__bf16)p1;
            float rsum = p0 + p1;
            rsum += __shfl_xor(rsum, 1);
            rsum += __shfl_xor(rsum, 2);
            rsum += __shfl_xor(rsum, 4);
            rsum += __shfl_xor(rsum, 8);
            l[r] = l[r] * al + rsum;
            o[0][r] *= al; o[1][r] *= al; o[2][r] *= al; o[3][r] *= al;
        }
        asm volatile("s_wait_dscnt 0" ::: "memory");   // same-wave LDS RAW guard

        // P (16x32) as A-fragment, V chunk (32x64) via vT as B-fragments
        v16bf pf = frag_ld(pl + l15 * 40, hi8);
        const __bf16* vrow = vT + ((size_t)(b * E + h * HS + l15)) * T + jj;
#pragma unroll
        for (int s = 0; s < 4; s++) {
            v16bf bv = frag_ld(vrow + (size_t)(s * 16) * T, hi8);
            o[s] = WMMA_BF16(pf, bv, o[s]);
        }
    }

#pragma unroll
    for (int s = 0; s < 4; s++) {
#pragma unroll
        for (int r = 0; r < 8; r++) {
            int row = qbase + r + hi8;
            att[((size_t)(b * T + row)) * E + h * HS + s * 16 + l15] = (__bf16)(o[s][r] / l[r]);
        }
    }
}

// ---------------------------------------------------------------------------
// Host orchestration
// ---------------------------------------------------------------------------
extern "C" void kernel_launch(void* const* d_in, const int* in_sizes, int n_in,
                              void* d_out, int out_size, void* d_ws, size_t ws_size,
                              hipStream_t stream) {
    constexpr int Bc = 4, Tc = 1024, Ec = 1024, Hc = 16, HSc = 64, Lc = 4, Fc = 4096, Vc = 32000;
    constexpr int Mc = Bc * Tc;   // 4096 rows
    (void)in_sizes; (void)n_in; (void)out_size; (void)ws_size;

    const int*   idx   = (const int*)  d_in[0];
    const float* tok   = (const float*)d_in[1];
    const float* pos   = (const float*)d_in[2];
    const float* Wq    = (const float*)d_in[3];
    const float* Wk    = (const float*)d_in[4];
    const float* Wv    = (const float*)d_in[5];
    const float* Wo    = (const float*)d_in[6];
    const float* bo    = (const float*)d_in[7];
    const float* ln1g  = (const float*)d_in[8];
    const float* ln1b  = (const float*)d_in[9];
    const float* ln2g  = (const float*)d_in[10];
    const float* ln2b  = (const float*)d_in[11];
    const float* W1    = (const float*)d_in[12];
    const float* b1    = (const float*)d_in[13];
    const float* W2    = (const float*)d_in[14];
    const float* b2    = (const float*)d_in[15];
    const float* lnfg  = (const float*)d_in[16];
    const float* lnfb  = (const float*)d_in[17];
    const float* Wh    = (const float*)d_in[18];
    const float* bh    = (const float*)d_in[19];
    float* logits = (float*)d_out;

    // Workspace carving (256B aligned)
    char* ws = (char*)d_ws;
    size_t off = 0;
    auto alloc = [&](size_t bytes) -> void* {
        void* p = ws + off;
        off += (bytes + 255) & ~(size_t)255;
        return p;
    };
    __bf16* WqT = (__bf16*)alloc((size_t)Lc * Ec * Ec * 2);
    __bf16* WkT = (__bf16*)alloc((size_t)Lc * Ec * Ec * 2);
    __bf16* WvT = (__bf16*)alloc((size_t)Lc * Ec * Ec * 2);
    __bf16* WoT = (__bf16*)alloc((size_t)Lc * Ec * Ec * 2);
    __bf16* W1T = (__bf16*)alloc((size_t)Lc * Ec * Fc * 2);
    __bf16* W2T = (__bf16*)alloc((size_t)Lc * Fc * Ec * 2);
    __bf16* WhT = (__bf16*)alloc((size_t)Vc * Ec * 2);
    float*  x   = (float*) alloc((size_t)Mc * Ec * 4);
    __bf16* hb  = (__bf16*)alloc((size_t)Mc * Ec * 2);
    __bf16* qb  = (__bf16*)alloc((size_t)Mc * Ec * 2);
    __bf16* kb  = (__bf16*)alloc((size_t)Mc * Ec * 2);
    __bf16* vb  = (__bf16*)alloc((size_t)Mc * Ec * 2);
    __bf16* vTb = (__bf16*)alloc((size_t)Bc * Ec * Tc * 2);
    __bf16* att = (__bf16*)alloc((size_t)Mc * Ec * 2);
    __bf16* ub  = (__bf16*)alloc((size_t)Mc * Fc * 2);

    auto cdiv = [](int a, int b) { return (a + b - 1) / b; };

    // Weight conversion (per-call: deterministic, graph-capture safe)
    {
        int tot = Lc * Ec * Ec;
        conv_qkvT_kernel<<<cdiv(tot, 256), 256, 0, stream>>>(Wq, WqT, tot);
        conv_qkvT_kernel<<<cdiv(tot, 256), 256, 0, stream>>>(Wk, WkT, tot);
        conv_qkvT_kernel<<<cdiv(tot, 256), 256, 0, stream>>>(Wv, WvT, tot);
        transpose_to_bf16_kernel<<<cdiv(tot, 256), 256, 0, stream>>>(Wo, WoT, Ec, Ec, tot);
        int tot1 = Lc * Ec * Fc;
        transpose_to_bf16_kernel<<<cdiv(tot1, 256), 256, 0, stream>>>(W1, W1T, Ec, Fc, tot1);
        transpose_to_bf16_kernel<<<cdiv(tot1, 256), 256, 0, stream>>>(W2, W2T, Fc, Ec, tot1);
        int toth = Ec * Vc;
        transpose_to_bf16_kernel<<<cdiv(toth, 256), 256, 0, stream>>>(Wh, WhT, Ec, Vc, toth);
    }

    // Embedding
    {
        int tot = Mc * Ec;
        embed_kernel<<<cdiv(tot, 256), 256, 0, stream>>>(idx, tok, pos, x, tot);
    }

    auto gemm = [&](const __bf16* A, const __bf16* Bt, const float* bias, const float* res,
                    float* Cf, __bf16* Cb, int N, int K, int relu) {
        dim3 grid(N / 64, Mc / 64);
        gemm_bf16_kernel<<<grid, 128, 0, stream>>>(A, Bt, bias, res, Cf, Cb, N, K, relu);
    };

    const float scale = 1.0f / 32.0f;   // 1/sqrt(E), matching the reference

    for (int l = 0; l < Lc; l++) {
        // h = LN1(x)
        layernorm_kernel<<<Mc, 256, 0, stream>>>(x, ln1g + (size_t)l * Ec, ln1b + (size_t)l * Ec, hb);
        // q,k,v = h @ W{q,k,v}
        gemm(hb, WqT + (size_t)l * Ec * Ec, nullptr, nullptr, nullptr, qb, Ec, Ec, 0);
        gemm(hb, WkT + (size_t)l * Ec * Ec, nullptr, nullptr, nullptr, kb, Ec, Ec, 0);
        gemm(hb, WvT + (size_t)l * Ec * Ec, nullptr, nullptr, nullptr, vb, Ec, Ec, 0);
        // vT for P.V fragment contiguity
        {
            int tot = Bc * Ec * Tc;
            vtrans_kernel<<<cdiv(tot, 256), 256, 0, stream>>>(vb, vTb, tot);
        }
        // flash attention
        {
            dim3 grid(Tc / 64, Hc, Bc);
            attn_kernel<<<grid, 128, 0, stream>>>(qb, kb, vTb, att, scale);
        }
        // x = x + att @ Wo + bo
        gemm(att, WoT + (size_t)l * Ec * Ec, bo + (size_t)l * Ec, x, x, nullptr, Ec, Ec, 0);
        // h = LN2(x)
        layernorm_kernel<<<Mc, 256, 0, stream>>>(x, ln2g + (size_t)l * Ec, ln2b + (size_t)l * Ec, hb);
        // u = relu(h @ W1 + b1)
        gemm(hb, W1T + (size_t)l * Ec * Fc, b1 + (size_t)l * Fc, nullptr, nullptr, ub, Fc, Ec, 1);
        // x = x + u @ W2 + b2
        gemm(ub, W2T + (size_t)l * Fc * Ec, b2 + (size_t)l * Ec, x, x, nullptr, Ec, Fc, 0);
    }

    // final LN + vocab head
    layernorm_kernel<<<Mc, 256, 0, stream>>>(x, lnfg, lnfb, hb);
    gemm(hb, WhT, bh, nullptr, logits, nullptr, Vc, Ec, 0);
}
